// MultiHeadAttention_16355235463422
// MI455X (gfx1250) — compile-verified, matrix operand order checked by reading, not run on hardware
//
#include <hip/hip_runtime.h>

typedef __attribute__((ext_vector_type(16))) _Float16 v16h;
typedef __attribute__((ext_vector_type(8)))  _Float16 v8h;
typedef __attribute__((ext_vector_type(8)))  float    v8f;

constexpr int Bc  = 4;
constexpr int Sc  = 2048;
constexpr int DMc = 512;   // model dim == H*Dh
constexpr int Hc  = 8;
constexpr int Dh  = 64;

// ---------------------------------------------------------------------------
// WMMA helpers (CDNA5 wave32, v_wmma_f32_16x16x32_f16)
// A-fragment (16x32 f16) per ISA 7.12.2: lane m = lane&15; lanes 0-15 hold
// K = {0..7, 16..23}, lanes 16-31 hold K = {8..15, 24..31}; i.e. two
// contiguous 8-half (16B) chunks at +kh*8 and +16+kh*8. B fragments use the
// mirrored layout (row index = N), so the same loader works for any matrix
// stored contiguous in the contraction dim with row stride `ld`.
// ---------------------------------------------------------------------------
__device__ __forceinline__ v16h load_frag(const _Float16* base, int ld) {
  const int lane = threadIdx.x & 31;
  const int m  = lane & 15;
  const int kh = (lane >> 4) & 1;
  union { v16h v; v8h h[2]; } u;
  const _Float16* p = base + m * ld + kh * 8;
  u.h[0] = *(const v8h*)(p);
  u.h[1] = *(const v8h*)(p + 16);
  return u.v;
}

__device__ __forceinline__ v8f wmma_f16(v16h a, v16h b, v8f c) {
  return __builtin_amdgcn_wmma_f32_16x16x32_f16(false, a, false, b,
                                                (short)0, c, false, false);
}

// ---------------------------------------------------------------------------
// fp32 -> f16 staging
// ---------------------------------------------------------------------------
__global__ __launch_bounds__(256) void cast_f32_to_f16(
    const float* __restrict__ src, _Float16* __restrict__ dst, int n) {
  int i = blockIdx.x * blockDim.x + threadIdx.x;
  int stride = gridDim.x * blockDim.x;
  for (; i < n; i += stride) dst[i] = (_Float16)src[i];
}

// ---------------------------------------------------------------------------
// Per-head projection: P[b,h,s,:] = X[b,s,:] @ W[h] + bias[h]
// grid = (BS/128, H, 3{q,k,v}); block = 256 (8 waves x 16 rows each)
// ---------------------------------------------------------------------------
__global__ __launch_bounds__(256) void proj_kernel(
    const _Float16* __restrict__ Xq, const _Float16* __restrict__ Xk,
    const _Float16* __restrict__ Xv,
    const _Float16* __restrict__ Wq16, const _Float16* __restrict__ Wk16,
    const _Float16* __restrict__ Wv16,
    const float* __restrict__ bq, const float* __restrict__ bk,
    const float* __restrict__ bv,
    _Float16* __restrict__ Qo, _Float16* __restrict__ Ko,
    _Float16* __restrict__ Vo) {
  __shared__ _Float16 Wt[Dh * DMc];  // 64KB: W[h] transposed -> Wt[n][k]

  const int z = blockIdx.z;
  const int h = blockIdx.y;
  const _Float16* X    = (z == 0) ? Xq   : (z == 1) ? Xk   : Xv;
  const _Float16* W    = (z == 0) ? Wq16 : (z == 1) ? Wk16 : Wv16;
  const float*    bias = (z == 0) ? bq   : (z == 1) ? bk   : bv;
  _Float16*       dst  = (z == 0) ? Qo   : (z == 1) ? Ko   : Vo;

  const _Float16* Wh = W + (size_t)h * DMc * Dh;
  for (int c = threadIdx.x; c < (DMc * Dh) / 8; c += 256) {
    const int k  = c >> 3;        // 0..511
    const int nb = (c & 7) * 8;   // 0..56
    v8h w = *(const v8h*)(Wh + k * Dh + nb);
#pragma unroll
    for (int t = 0; t < 8; ++t) Wt[(nb + t) * DMc + k] = w[t];
  }
  __syncthreads();

  const int wave = threadIdx.x >> 5;
  const int lane = threadIdx.x & 31;
  const int col  = lane & 15;
  const int half = (lane >> 4) & 1;
  const int r0   = blockIdx.x * 128 + wave * 16;  // global row in [0, B*S)

  v8f acc[4];
#pragma unroll
  for (int nt = 0; nt < 4; ++nt) {
    float bb = bias[h * Dh + nt * 16 + col];
#pragma unroll
    for (int j = 0; j < 8; ++j) acc[nt][j] = bb;
  }

  const _Float16* Xrow = X + (size_t)r0 * DMc;
  for (int k0 = 0; k0 < DMc; k0 += 32) {
    v16h a = load_frag(Xrow + k0, DMc);
#pragma unroll
    for (int nt = 0; nt < 4; ++nt) {
      v16h b = load_frag(Wt + (nt * 16) * DMc + k0, DMc);
      acc[nt] = wmma_f16(a, b, acc[nt]);
    }
  }

#pragma unroll
  for (int nt = 0; nt < 4; ++nt) {
#pragma unroll
    for (int j = 0; j < 8; ++j) {
      int rg = r0 + j + 8 * half;
      int b  = rg >> 11;          // / S
      int s  = rg & (Sc - 1);
      dst[(((size_t)b * Hc + h) * Sc + s) * Dh + nt * 16 + col] =
          (_Float16)acc[nt][j];
    }
  }
}

// ---------------------------------------------------------------------------
// Flash attention over one (b,h): grid = (S/128, B*H), block = 256.
// Each wave owns 16 query rows; block streams 32-key tiles through LDS.
// ---------------------------------------------------------------------------
__global__ __launch_bounds__(256) void attn_kernel(
    const _Float16* __restrict__ Q, const _Float16* __restrict__ K,
    const _Float16* __restrict__ V, _Float16* __restrict__ Cc) {
  __shared__ _Float16 Ks[32 * Dh];       // K tile   [key][d]   4KB
  __shared__ _Float16 Vt[Dh * 32];       // V tile^T [d][key]   4KB
  __shared__ _Float16 Ps[8 * 16 * 32];   // per-wave P staging  8KB

  const int bh   = blockIdx.y;
  const int b    = bh / Hc;
  const int h    = bh % Hc;
  const int wave = threadIdx.x >> 5;
  const int lane = threadIdx.x & 31;
  const int col  = lane & 15;
  const int half = (lane >> 4) & 1;
  const int qr0  = blockIdx.x * 128 + wave * 16;   // query row within S

  const _Float16* Qg = Q + (size_t)bh * Sc * Dh;
  const _Float16* Kg = K + (size_t)bh * Sc * Dh;
  const _Float16* Vg = V + (size_t)bh * Sc * Dh;
  _Float16* Pw = Ps + wave * (16 * 32);

  const v16h qa0 = load_frag(Qg + (size_t)qr0 * Dh + 0, Dh);
  const v16h qa1 = load_frag(Qg + (size_t)qr0 * Dh + 32, Dh);

  float mrow[8], lrow[8];
  v8f o[4];
#pragma unroll
  for (int j = 0; j < 8; ++j) { mrow[j] = -3.0e38f; lrow[j] = 0.0f; }
#pragma unroll
  for (int nt = 0; nt < 4; ++nt)
#pragma unroll
    for (int j = 0; j < 8; ++j) o[nt][j] = 0.0f;

  const float scale = 0.125f;  // 1/sqrt(64)

  for (int t0 = 0; t0 < Sc; t0 += 32) {
    // --- cooperative stage: K tile straight, V tile transposed ---
    for (int c = threadIdx.x; c < (32 * Dh) / 8; c += 256)
      ((v8h*)Ks)[c] = ((const v8h*)(Kg + (size_t)t0 * Dh))[c];
    for (int c = threadIdx.x; c < (32 * Dh) / 8; c += 256) {
      const int key = c >> 3;
      const int d0  = (c & 7) * 8;
      v8h vv = *(const v8h*)(Vg + (size_t)(t0 + key) * Dh + d0);
#pragma unroll
      for (int t = 0; t < 8; ++t) Vt[(d0 + t) * 32 + key] = vv[t];
    }
    if (t0 + 32 < Sc) {
      __builtin_prefetch(Kg + (size_t)(t0 + 32) * Dh + threadIdx.x * 8, 0, 1);
      __builtin_prefetch(Vg + (size_t)(t0 + 32) * Dh + threadIdx.x * 8, 0, 1);
    }
    __syncthreads();

    // --- scores: S(16x32) = Qtile(16x64) . Ktile^T ---
    v8f s0, s1;
#pragma unroll
    for (int j = 0; j < 8; ++j) { s0[j] = 0.0f; s1[j] = 0.0f; }
    s0 = wmma_f16(qa0, load_frag(Ks + 0, Dh), s0);
    s0 = wmma_f16(qa1, load_frag(Ks + 32, Dh), s0);
    s1 = wmma_f16(qa0, load_frag(Ks + 16 * Dh + 0, Dh), s1);
    s1 = wmma_f16(qa1, load_frag(Ks + 16 * Dh + 32, Dh), s1);

    // --- online softmax (rows live in 16-lane halves of the C layout) ---
#pragma unroll
    for (int j = 0; j < 8; ++j) {
      float x0 = s0[j] * scale;
      float x1 = s1[j] * scale;
      float rm = fmaxf(x0, x1);
      rm = fmaxf(rm, __shfl_xor(rm, 1, 32));
      rm = fmaxf(rm, __shfl_xor(rm, 2, 32));
      rm = fmaxf(rm, __shfl_xor(rm, 4, 32));
      rm = fmaxf(rm, __shfl_xor(rm, 8, 32));
      float mnew = fmaxf(mrow[j], rm);
      float corr = __expf(mrow[j] - mnew);
      float p0 = __expf(x0 - mnew);
      float p1 = __expf(x1 - mnew);
      float rs = p0 + p1;
      rs += __shfl_xor(rs, 1, 32);
      rs += __shfl_xor(rs, 2, 32);
      rs += __shfl_xor(rs, 4, 32);
      rs += __shfl_xor(rs, 8, 32);
      lrow[j] = lrow[j] * corr + rs;
      mrow[j] = mnew;
      o[0][j] *= corr; o[1][j] *= corr; o[2][j] *= corr; o[3][j] *= corr;
      // C-layout -> A-layout via wave-private LDS (same-wave DS is in-order)
      const int prow = j + 8 * half;
      Pw[prow * 32 + col]      = (_Float16)p0;
      Pw[prow * 32 + 16 + col] = (_Float16)p1;
    }

    // --- O += P(16x32) . Vtile(32x64) ---
    v16h pa = load_frag(Pw, 32);
#pragma unroll
    for (int nt = 0; nt < 4; ++nt) {
      v16h vb = load_frag(Vt + (nt * 16) * 32, 32);
      o[nt] = wmma_f16(pa, vb, o[nt]);
    }
    __syncthreads();
  }

  // --- normalize and scatter into concat buffer [B,S,H*Dh] ---
  float inv[8];
#pragma unroll
  for (int j = 0; j < 8; ++j) inv[j] = 1.0f / lrow[j];
#pragma unroll
  for (int nt = 0; nt < 4; ++nt) {
#pragma unroll
    for (int j = 0; j < 8; ++j) {
      int r = qr0 + j + 8 * half;
      Cc[((size_t)b * Sc + r) * DMc + h * Dh + nt * 16 + col] =
          (_Float16)(o[nt][j] * inv[j]);
    }
  }
}

// ---------------------------------------------------------------------------
// Output projection: out[BS,512] = Cc[BS,512] @ Wo[512,512] + bo (fp32 out)
// grid = (BS/128, 512/64); block = 256
// ---------------------------------------------------------------------------
__global__ __launch_bounds__(256) void outproj_kernel(
    const _Float16* __restrict__ Cc, const _Float16* __restrict__ Wo16,
    const float* __restrict__ bo, float* __restrict__ out) {
  __shared__ _Float16 Wot[64 * DMc];  // 64KB: Wo[:, n0:n0+64] transposed

  const int n0 = blockIdx.y * 64;
  for (int c = threadIdx.x; c < (DMc * 64) / 8; c += 256) {
    const int k  = c >> 3;
    const int nb = (c & 7) * 8;
    v8h w = *(const v8h*)(Wo16 + (size_t)k * DMc + n0 + nb);
#pragma unroll
    for (int t = 0; t < 8; ++t) Wot[(nb + t) * DMc + k] = w[t];
  }
  __syncthreads();

  const int wave = threadIdx.x >> 5;
  const int lane = threadIdx.x & 31;
  const int col  = lane & 15;
  const int half = (lane >> 4) & 1;
  const int r0   = blockIdx.x * 128 + wave * 16;

  v8f acc[4];
#pragma unroll
  for (int nt = 0; nt < 4; ++nt) {
    float bb = bo[n0 + nt * 16 + col];
#pragma unroll
    for (int j = 0; j < 8; ++j) acc[nt][j] = bb;
  }

  const _Float16* Crow = Cc + (size_t)r0 * DMc;
  for (int k0 = 0; k0 < DMc; k0 += 32) {
    v16h a = load_frag(Crow + k0, DMc);
#pragma unroll
    for (int nt = 0; nt < 4; ++nt) {
      v16h b = load_frag(Wot + (nt * 16) * DMc + k0, DMc);
      acc[nt] = wmma_f16(a, b, acc[nt]);
    }
  }

#pragma unroll
  for (int nt = 0; nt < 4; ++nt) {
#pragma unroll
    for (int j = 0; j < 8; ++j) {
      int rg = r0 + j + 8 * half;
      out[(size_t)rg * DMc + n0 + nt * 16 + col] = acc[nt][j];
    }
  }
}

// ---------------------------------------------------------------------------
extern "C" void kernel_launch(void* const* d_in, const int* in_sizes, int n_in,
                              void* d_out, int out_size, void* d_ws,
                              size_t ws_size, hipStream_t stream) {
  (void)in_sizes; (void)n_in; (void)out_size; (void)ws_size;
  const float* q_in = (const float*)d_in[0];
  const float* k_in = (const float*)d_in[1];
  const float* v_in = (const float*)d_in[2];
  const float* Wq   = (const float*)d_in[3];
  const float* bq   = (const float*)d_in[4];
  const float* Wk   = (const float*)d_in[5];
  const float* bk   = (const float*)d_in[6];
  const float* Wv   = (const float*)d_in[7];
  const float* bv   = (const float*)d_in[8];
  const float* Wo   = (const float*)d_in[9];
  const float* bo   = (const float*)d_in[10];
  float* out = (float*)d_out;

  // workspace layout (f16 elements), total ~58 MB
  const size_t NX  = (size_t)Bc * Sc * DMc;     // 4,194,304
  const size_t NW  = (size_t)Hc * DMc * Dh;     //   262,144
  const size_t NWo = (size_t)DMc * DMc;         //   262,144
  const size_t NP  = (size_t)Bc * Hc * Sc * Dh; // 4,194,304
  _Float16* ws   = (_Float16*)d_ws;
  _Float16* Xq   = ws;
  _Float16* Xk   = Xq + NX;
  _Float16* Xv   = Xk + NX;
  _Float16* Wq16 = Xv + NX;
  _Float16* Wk16 = Wq16 + NW;
  _Float16* Wv16 = Wk16 + NW;
  _Float16* Wo16 = Wv16 + NW;
  _Float16* Qp   = Wo16 + NWo;
  _Float16* Kp   = Qp + NP;
  _Float16* Vp   = Kp + NP;
  _Float16* Cc   = Vp + NP;

  cast_f32_to_f16<<<2048, 256, 0, stream>>>(q_in, Xq, (int)NX);
  cast_f32_to_f16<<<2048, 256, 0, stream>>>(k_in, Xk, (int)NX);
  cast_f32_to_f16<<<2048, 256, 0, stream>>>(v_in, Xv, (int)NX);
  cast_f32_to_f16<<<256, 256, 0, stream>>>(Wq, Wq16, (int)NW);
  cast_f32_to_f16<<<256, 256, 0, stream>>>(Wk, Wk16, (int)NW);
  cast_f32_to_f16<<<256, 256, 0, stream>>>(Wv, Wv16, (int)NW);
  cast_f32_to_f16<<<256, 256, 0, stream>>>(Wo, Wo16, (int)NWo);

  proj_kernel<<<dim3((Bc * Sc) / 128, Hc, 3), 256, 0, stream>>>(
      Xq, Xk, Xv, Wq16, Wk16, Wv16, bq, bk, bv, Qp, Kp, Vp);

  attn_kernel<<<dim3(Sc / 128, Bc * Hc), 256, 0, stream>>>(Qp, Kp, Vp, Cc);

  outproj_kernel<<<dim3((Bc * Sc) / 128, DMc / 64), 256, 0, stream>>>(
      Cc, Wo16, bo, out);
}
